// BiSTFGRN_70944269796066
// MI455X (gfx1250) — compile-verified, matrix operand order checked by reading, not run on hardware
//
#include <hip/hip_runtime.h>
#include <hip/hip_bf16.h>
#include <stdint.h>

// ---------------- problem constants ----------------
#define NN    883          // nodes
#define TT    12           // time steps
#define BB    64           // batch
#define DIN   2
#define DOUT  64
#define DE    10
#define DI    66           // DIN + DOUT
#define GO    128          // 2*DOUT (gate output)
#define BDI   (BB*DI)      // 4224 columns of S@inp GEMM
#define NPAD  896          // 883 padded to 16/32 multiple
#define KP    160          // 2*DI=132 padded to 32 multiple
#define KTILES (NPAD/32)   // 28 K-steps in the supports GEMM

#if __has_builtin(__builtin_amdgcn_global_load_async_to_lds_b128) && \
    __has_builtin(__builtin_amdgcn_s_wait_asynccnt)
#define HAVE_ASYNC_LDS 1
#else
#define HAVE_ASYNC_LDS 0
#endif

typedef __attribute__((ext_vector_type(16))) _Float16 v16h;
typedef __attribute__((ext_vector_type(8)))  float    v8f;
typedef __attribute__((ext_vector_type(4)))  int      v4i;

#if HAVE_ASYNC_LDS
typedef __attribute__((address_space(1))) v4i* gv4p;   // global int4*
typedef __attribute__((address_space(3))) v4i* lv4p;   // LDS int4*
#endif

union frag16 { v16h h; uint32_t u[8]; };

// ---- WMMA: D = A(16x32 f16) * B(32x16 f16) + C(16x16 f32) ----
__device__ __forceinline__ v8f wmma32(v16h a, v16h b, v8f c) {
  return __builtin_amdgcn_wmma_f32_16x16x32_f16(
      /*neg_a=*/false, a, /*neg_b=*/false, b,
      /*c_mod=*/(short)0, c, /*reuse_a=*/false, /*reuse_b=*/false);
}

// Fragment loader for a row-major [rows][ld] f16 matrix: row index striped over
// lanes, K contiguous. Matches CDNA5 16-bit A (M=lane&15) / B ([N][K] storage):
//   lane l: row = row0 + (l&15); khalf = (l>>4)*8
//   VGPR v: k = k0 + khalf + (v>>2)*16 + (v&3)*2   (packed f16 pair)
__device__ __forceinline__ v16h load_frag(const _Float16* __restrict__ base,
                                          int ld, int row0, int k0, int lane) {
  frag16 f;
  const _Float16* p = base + (size_t)(row0 + (lane & 15)) * (size_t)ld
                      + k0 + ((lane >> 4) & 1) * 8;
#pragma unroll
  for (int v = 0; v < 8; ++v) {
    int off = ((v >> 2) * 16) + ((v & 3) * 2);
    f.u[v] = *reinterpret_cast<const uint32_t*>(p + off);
  }
  return f.h;
}

// Same fragment pattern but from an LDS-staged tile [rows][32] (k0 == 0).
__device__ __forceinline__ v16h load_frag_lds(const _Float16* base,
                                              int row0, int lane) {
  frag16 f;
  const _Float16* p = base + (row0 + (lane & 15)) * 32 + ((lane >> 4) & 1) * 8;
#pragma unroll
  for (int v = 0; v < 8; ++v)
    f.u[v] = *reinterpret_cast<const uint32_t*>(p + ((v >> 2) * 16) + ((v & 3) * 2));
  return f.h;
}

// A-fragment for the per-node GEMMs: logical A row b, K = [inp (66) | xg (66) | pad].
// Both regions are laid out as  base + b*66 + i  (node-major buffers).
__device__ __forceinline__ v16h load_frag_concat(const _Float16* __restrict__ p1,
                                                 const _Float16* __restrict__ p2,
                                                 int b0, int k0, int lane) {
  frag16 f;
  size_t r66 = (size_t)(b0 + (lane & 15)) * DI;
  int kh = ((lane >> 4) & 1) * 8;
#pragma unroll
  for (int v = 0; v < 8; ++v) {
    int k = k0 + kh + ((v >> 2) * 16) + ((v & 3) * 2);
    uint32_t u = 0u;
    if (k < DI)            u = *reinterpret_cast<const uint32_t*>(p1 + r66 + k);
    else if (k < 2 * DI)   u = *reinterpret_cast<const uint32_t*>(p2 + r66 + (k - DI));
    f.u[v] = u;
  }
  return f.h;
}

// ---------------- precompute: S = softmax(relu(E E^T)) + adj (f16, padded) ---
__global__ void k_supports(const float* __restrict__ E,
                           const float* __restrict__ adj,
                           _Float16* __restrict__ S) {
  int n = blockIdx.x;
  if (n >= NN) {  // zero pad rows 883..895
    for (int m = threadIdx.x; m < NPAD; m += blockDim.x)
      S[(size_t)n * NPAD + m] = (_Float16)0.f;
    return;
  }
  __shared__ float en[DE];
  __shared__ float red[256];
  if (threadIdx.x < DE) en[threadIdx.x] = E[n * DE + threadIdx.x];
  __syncthreads();

  float mx = 0.f;  // relu >= 0
  for (int m = threadIdx.x; m < NN; m += 256) {
    float d = 0.f;
#pragma unroll
    for (int dd = 0; dd < DE; ++dd) d += en[dd] * E[m * DE + dd];
    mx = fmaxf(mx, fmaxf(d, 0.f));
  }
  red[threadIdx.x] = mx; __syncthreads();
  for (int s = 128; s > 0; s >>= 1) {
    if (threadIdx.x < s) red[threadIdx.x] = fmaxf(red[threadIdx.x], red[threadIdx.x + s]);
    __syncthreads();
  }
  mx = red[0]; __syncthreads();

  float sum = 0.f;
  for (int m = threadIdx.x; m < NN; m += 256) {
    float d = 0.f;
#pragma unroll
    for (int dd = 0; dd < DE; ++dd) d += en[dd] * E[m * DE + dd];
    sum += expf(fmaxf(d, 0.f) - mx);
  }
  red[threadIdx.x] = sum; __syncthreads();
  for (int s = 128; s > 0; s >>= 1) {
    if (threadIdx.x < s) red[threadIdx.x] += red[threadIdx.x + s];
    __syncthreads();
  }
  float inv = 1.f / red[0]; __syncthreads();

  for (int m = threadIdx.x; m < NN; m += 256) {
    float d = 0.f;
#pragma unroll
    for (int dd = 0; dd < DE; ++dd) d += en[dd] * E[m * DE + dd];
    float a = expf(fmaxf(d, 0.f) - mx) * inv;
    S[(size_t)n * NPAD + m] = (_Float16)(a + adj[(size_t)n * NN + m]);
  }
  for (int m = NN + threadIdx.x; m < NPAD; m += 256)
    S[(size_t)n * NPAD + m] = (_Float16)0.f;
}

// --- per-node weights: outW[n][o][k] f16 (k padded to KP), outB[n][o] f32 ---
__global__ void k_mkweights(const float* __restrict__ E,
                            const float* __restrict__ W,   // [DE][2][DI][O]
                            const float* __restrict__ bsrc,// [DE][O]
                            _Float16* __restrict__ outW,
                            float* __restrict__ outB, int O) {
  int n = blockIdx.x;
  float en[DE];
#pragma unroll
  for (int d = 0; d < DE; ++d) en[d] = E[n * DE + d];
  const int total = O * KP;
  const size_t dstride = (size_t)2 * DI * O;
  for (int idx = threadIdx.x; idx < total; idx += blockDim.x) {
    int o = idx / KP, k = idx - o * KP;
    float v = 0.f;
    if (k < 2 * DI) {
      int kk = (k < DI) ? 0 : 1;
      int ii = (k < DI) ? k : k - DI;
      const float* wp = W + ((size_t)kk * DI + ii) * O + o;
#pragma unroll
      for (int d = 0; d < DE; ++d) v += en[d] * wp[d * dstride];
    }
    outW[(size_t)n * total + idx] = (_Float16)v;
  }
  for (int o = threadIdx.x; o < O; o += blockDim.x) {
    float v = 0.f;
#pragma unroll
    for (int d = 0; d < DE; ++d) v += en[d] * bsrc[d * O + o];
    outB[(size_t)n * O + o] = v;
  }
}

__global__ void k_zero_f32(float* __restrict__ p, int count) {
  int i = blockIdx.x * blockDim.x + threadIdx.x;
  if (i < count) p[i] = 0.f;
}

// -------- per step: inp = concat(x_t, h) in node-major and col-major --------
__global__ void k_build_inp(const float* __restrict__ x, const float* __restrict__ h,
                            int tx,
                            _Float16* __restrict__ inp_node,
                            _Float16* __restrict__ inp_col,
                            _Float16* __restrict__ inp2_node,
                            _Float16* __restrict__ inp2_col) {
  int n = blockIdx.x;
  if (n >= NN) {  // zero pad K-rows of the column-major B operands
    for (int c = threadIdx.x; c < BDI; c += blockDim.x) {
      inp_col[(size_t)c * NPAD + n]  = (_Float16)0.f;
      inp2_col[(size_t)c * NPAD + n] = (_Float16)0.f;
    }
    return;
  }
  for (int c = threadIdx.x; c < BDI; c += blockDim.x) {
    int b = c / DI, i = c - b * DI;
    float v = (i < DIN)
        ? x[(((size_t)b * TT + tx) * NN + n) * DIN + i]
        : h[((size_t)n * BB + b) * DOUT + (i - DIN)];
    _Float16 hv = (_Float16)v;
    inp_node[(size_t)n * BDI + c] = hv;
    inp_col[(size_t)c * NPAD + n] = hv;
    if (i < DIN) {  // x-part of inp2 is identical; r*h part filled by k_gates
      inp2_node[(size_t)n * BDI + c] = hv;
      inp2_col[(size_t)c * NPAD + n] = hv;
    }
  }
}

// ---------------- GEMM: Xg[n][col] = sum_m S[n][m] * inp[m][col] ------------
// grid (14, 66), block 128 (4 waves). Wave computes a 16x64 strip.
// The 64x32 B tile is shared by all 4 waves: staged in LDS via async
// global->LDS b128 loads (double buffered, ASYNCcnt-tracked), A streamed
// from global. Xg is padded to NPAD rows so the epilogue is branch-free.
__global__ void __launch_bounds__(128)
k_gemm_supports(const _Float16* __restrict__ S,
                const _Float16* __restrict__ Bcol,   // [col][NPAD]
                _Float16* __restrict__ Xg) {         // [NPAD][BDI]
  __shared__ _Float16 bstage[2][64 * 32];            // 2 x 4KB
  const int tid = threadIdx.x;
  const int wave = tid >> 5, lane = tid & 31;
  const int mt = blockIdx.x * 4 + wave;              // 0..55
  const int col0 = blockIdx.y * 64;

  // fill assignment: each thread owns 32B of the 4KB tile (2 x b128)
  const int frow = tid >> 1, fhalf = tid & 1;        // row 0..63, 32B half
  const _Float16* gsrc = Bcol + (size_t)(col0 + frow) * NPAD + fhalf * 16;
  _Float16* lbase0 = &bstage[0][frow * 32 + fhalf * 16];
  _Float16* lbase1 = &bstage[1][frow * 32 + fhalf * 16];

  auto issue = [&](int kt, int sel) {
    const _Float16* g = gsrc + kt * 32;
    _Float16* l = sel ? lbase1 : lbase0;
#if HAVE_ASYNC_LDS
    __builtin_amdgcn_global_load_async_to_lds_b128((gv4p)(g),     (lv4p)(l),     0, 0);
    __builtin_amdgcn_global_load_async_to_lds_b128((gv4p)(g + 8), (lv4p)(l + 8), 0, 0);
#else
    *reinterpret_cast<uint4*>(l)     = *reinterpret_cast<const uint4*>(g);
    *reinterpret_cast<uint4*>(l + 8) = *reinterpret_cast<const uint4*>(g + 8);
#endif
  };

  v8f acc[4];
#pragma unroll
  for (int j = 0; j < 4; ++j) acc[j] = 0.f;

  issue(0, 0);
  for (int kt = 0; kt < KTILES; ++kt) {
    const int sel = kt & 1;
    if (kt + 1 < KTILES) {
      issue(kt + 1, sel ^ 1);                      // prefetch next tile
#if HAVE_ASYNC_LDS
      __builtin_amdgcn_s_wait_asynccnt(2);         // cur fills done, next pending
#endif
    } else {
#if HAVE_ASYNC_LDS
      __builtin_amdgcn_s_wait_asynccnt(0);         // last tile: drain
#endif
    }
    __syncthreads();                               // all waves' fills landed
    v16h a = load_frag(S, NPAD, mt * 16, kt * 32, lane);
#pragma unroll
    for (int j = 0; j < 4; ++j) {
      v16h b = load_frag_lds(&bstage[sel][0], j * 16, lane);
      acc[j] = wmma32(a, b, acc[j]);
    }
    __syncthreads();                               // reads done before refill
  }

  const int nl = lane & 15, mh = (lane >> 4) * 8;
#pragma unroll
  for (int j = 0; j < 4; ++j)
#pragma unroll
    for (int r = 0; r < 8; ++r) {
      int m = mt * 16 + mh + r;                    // Xg padded: no guard
      Xg[(size_t)m * BDI + col0 + j * 16 + nl] = (_Float16)acc[j][r];
    }
}

// ------ gates: zr = sigmoid([inp|xg] @ Wg_n + bg_n); emit z and inp2 --------
// grid NN, block 256 (8 waves). Wave = (b-tile, 4 o-tiles) of the 64x128 out.
__global__ void __launch_bounds__(256)
k_gates(const _Float16* __restrict__ inp_node,
        const _Float16* __restrict__ xg,
        const _Float16* __restrict__ WgN,  // [NN][GO][KP]
        const float* __restrict__ bgN,     // [NN][GO]
        const float* __restrict__ h,       // [NN][BB][DOUT]
        float* __restrict__ z,             // [NN][BB][DOUT]
        _Float16* __restrict__ inp2_node,
        _Float16* __restrict__ inp2_col) {
  int n = blockIdx.x;
  int wave = threadIdx.x >> 5, lane = threadIdx.x & 31;
  int mt = wave & 3, nt0 = (wave >> 2) * 4;
  const _Float16* p1 = inp_node + (size_t)n * BDI;
  const _Float16* p2 = xg + (size_t)n * BDI;
  const _Float16* wb = WgN + (size_t)n * GO * KP;
  v8f acc[4];
#pragma unroll
  for (int j = 0; j < 4; ++j) acc[j] = 0.f;
#pragma unroll
  for (int kt = 0; kt < KP / 32; ++kt) {
    v16h a = load_frag_concat(p1, p2, mt * 16, kt * 32, lane);
#pragma unroll
    for (int j = 0; j < 4; ++j) {
      v16h b = load_frag(wb, KP, (nt0 + j) * 16, kt * 32, lane);
      acc[j] = wmma32(a, b, acc[j]);
    }
  }
  int nl = lane & 15, mh = (lane >> 4) * 8;
#pragma unroll
  for (int j = 0; j < 4; ++j)
#pragma unroll
    for (int r = 0; r < 8; ++r) {
      int b = mt * 16 + mh + r;
      int o = (nt0 + j) * 16 + nl;
      float v = acc[j][r] + bgN[(size_t)n * GO + o];
      float s = 1.f / (1.f + expf(-v));
      if (o < DOUT) {
        z[((size_t)n * BB + b) * DOUT + o] = s;
      } else {
        int oo = o - DOUT;
        float rh = s * h[((size_t)n * BB + b) * DOUT + oo];
        _Float16 hv = (_Float16)rh;
        inp2_node[(size_t)n * BDI + b * DI + DIN + oo] = hv;
        inp2_col[((size_t)b * DI + DIN + oo) * NPAD + n] = hv;
      }
    }
}

// ---- candidate: hc = tanh([inp2|xg2] @ Wc_n + bc_n); GRU update + output ---
__global__ void __launch_bounds__(256)
k_cand(const _Float16* __restrict__ inp2_node,
       const _Float16* __restrict__ xg,
       const _Float16* __restrict__ WcN,  // [NN][DOUT][KP]
       const float* __restrict__ bcN,     // [NN][DOUT]
       const float* __restrict__ z,
       float* __restrict__ h,
       float* __restrict__ out, int step, int chan0) {
  int n = blockIdx.x;
  int wave = threadIdx.x >> 5, lane = threadIdx.x & 31;
  int mt = wave & 3, nt0 = (wave >> 2) * 2;
  const _Float16* p1 = inp2_node + (size_t)n * BDI;
  const _Float16* p2 = xg + (size_t)n * BDI;
  const _Float16* wb = WcN + (size_t)n * DOUT * KP;
  v8f acc[2];
#pragma unroll
  for (int j = 0; j < 2; ++j) acc[j] = 0.f;
#pragma unroll
  for (int kt = 0; kt < KP / 32; ++kt) {
    v16h a = load_frag_concat(p1, p2, mt * 16, kt * 32, lane);
#pragma unroll
    for (int j = 0; j < 2; ++j) {
      v16h b = load_frag(wb, KP, (nt0 + j) * 16, kt * 32, lane);
      acc[j] = wmma32(a, b, acc[j]);
    }
  }
  int nl = lane & 15, mh = (lane >> 4) * 8;
#pragma unroll
  for (int j = 0; j < 2; ++j)
#pragma unroll
    for (int r = 0; r < 8; ++r) {
      int b = mt * 16 + mh + r;
      int o = (nt0 + j) * 16 + nl;
      float hc = tanhf(acc[j][r] + bcN[(size_t)n * DOUT + o]);
      size_t hi = ((size_t)n * BB + b) * DOUT + o;
      float zz = z[hi];
      float hn = zz * h[hi] + (1.f - zz) * hc;
      h[hi] = hn;
      out[(((size_t)b * TT + step) * NN + n) * (2 * DOUT) + chan0 + o] = hn;
    }
}

// ------------------------------- host side ----------------------------------
extern "C" void kernel_launch(void* const* d_in, const int* in_sizes, int n_in,
                              void* d_out, int out_size, void* d_ws, size_t ws_size,
                              hipStream_t stream) {
  const float* x    = (const float*)d_in[0];
  const float* adj  = (const float*)d_in[1];
  const float* E    = (const float*)d_in[2];
  const float* Wg_f = (const float*)d_in[3];
  const float* bg_f = (const float*)d_in[4];
  const float* Wc_f = (const float*)d_in[5];
  const float* bc_f = (const float*)d_in[6];
  const float* Wg_b = (const float*)d_in[7];
  const float* bg_b = (const float*)d_in[8];
  const float* Wc_b = (const float*)d_in[9];
  const float* bc_b = (const float*)d_in[10];
  float* out = (float*)d_out;

  char* ws = (char*)d_ws;
  size_t off = 0;
  auto alloc = [&](size_t bytes) -> void* {
    off = (off + 255) & ~(size_t)255;
    void* p = ws + off;
    off += bytes;
    return p;
  };

  _Float16* S = (_Float16*)alloc((size_t)NPAD * NPAD * 2);
  _Float16* WgN[2], *WcN[2];
  float *bgN[2], *bcN[2];
  for (int d = 0; d < 2; ++d) {
    WgN[d] = (_Float16*)alloc((size_t)NN * GO * KP * 2);
    WcN[d] = (_Float16*)alloc((size_t)NN * DOUT * KP * 2);
    bgN[d] = (float*)alloc((size_t)NN * GO * 4);
    bcN[d] = (float*)alloc((size_t)NN * DOUT * 4);
  }
  _Float16* inp_node  = (_Float16*)alloc((size_t)NN * BDI * 2);
  _Float16* inp2_node = (_Float16*)alloc((size_t)NN * BDI * 2);
  _Float16* inp_col   = (_Float16*)alloc((size_t)BDI * NPAD * 2);
  _Float16* inp2_col  = (_Float16*)alloc((size_t)BDI * NPAD * 2);
  _Float16* xg        = (_Float16*)alloc((size_t)NPAD * BDI * 2);  // padded rows
  float* z            = (float*)alloc((size_t)NN * BB * DOUT * 4);
  float* h            = (float*)alloc((size_t)NN * BB * DOUT * 4);

  // -------- one-time precompute --------
  k_supports<<<NPAD, 256, 0, stream>>>(E, adj, S);
  k_mkweights<<<NN, 256, 0, stream>>>(E, Wg_f, bg_f, WgN[0], bgN[0], GO);
  k_mkweights<<<NN, 256, 0, stream>>>(E, Wc_f, bc_f, WcN[0], bcN[0], DOUT);
  k_mkweights<<<NN, 256, 0, stream>>>(E, Wg_b, bg_b, WgN[1], bgN[1], GO);
  k_mkweights<<<NN, 256, 0, stream>>>(E, Wc_b, bc_b, WcN[1], bcN[1], DOUT);

  const int hcount = NN * BB * DOUT;
  const dim3 gemm_grid(14, BDI / 64);  // 56 M-tiles / 4 waves, 66 column strips

  for (int dir = 0; dir < 2; ++dir) {
    k_zero_f32<<<(hcount + 255) / 256, 256, 0, stream>>>(h, hcount);
    for (int s = 0; s < TT; ++s) {
      int tx = dir ? (TT - 1 - s) : s;
      k_build_inp<<<NPAD, 256, 0, stream>>>(x, h, tx, inp_node, inp_col,
                                            inp2_node, inp2_col);
      k_gemm_supports<<<gemm_grid, 128, 0, stream>>>(S, inp_col, xg);
      k_gates<<<NN, 256, 0, stream>>>(inp_node, xg, WgN[dir], bgN[dir], h, z,
                                      inp2_node, inp2_col);
      k_gemm_supports<<<gemm_grid, 128, 0, stream>>>(S, inp2_col, xg);
      k_cand<<<NN, 256, 0, stream>>>(inp2_node, xg, WcN[dir], bcN[dir], z, h,
                                     out, s, dir * DOUT);
    }
  }
}
